// ShiftedWindowAttention_88441966559560
// MI455X (gfx1250) — matrix operand order repaired and checked by reading, not compiled, not run on hardware
//
#include <hip/hip_runtime.h>
#include <hip/hip_bf16.h>

typedef __attribute__((ext_vector_type(16))) _Float16 v16h;
typedef __attribute__((ext_vector_type(8)))  _Float16 v8h;
typedef __attribute__((ext_vector_type(8)))  float    v8f;

namespace {
constexpr int kDim   = 256;
constexpr int kHeads = 8;
constexpr int kShift = 4;
constexpr float kScale = 0.17677669529663687f;   // 32^-0.5

// LDS layout (bytes)
constexpr int OFF_X  = 0;        // 64 x 256 f16  = 32768
constexpr int OFF_Q  = 32768;    // 64 x 256 f16
constexpr int OFF_K  = 65536;    // 64 x 256 f16
constexpr int OFF_VT = 98304;    // 8h x 32d x 64t f16 (transposed V)
constexpr int OFF_S  = 131072;   // 8h x 64x64 f32 = 131072; first 64KB doubles
                                 // as f32 staging buffer for the async x load
constexpr int OFF_O  = 262144;   // 64 x 256 f16
constexpr int OFF_RS = 294912;   // 8h x 64 f32 row sums
constexpr int OFF_PE = 296960;   // 8 x 225 f32 pos_enc (7200 B)
constexpr int OFF_RG = 304160;   // 64 x i32 region labels
constexpr int kSmemBytes = 304416;
}

__device__ __forceinline__ v16h ld_frag(const _Float16* p0, const _Float16* p1) {
  union { v16h v; v8h h[2]; } u;
  u.h[0] = *(const v8h*)p0;   // fragment elems 0..7  (VGPR 0..3)
  u.h[1] = *(const v8h*)p1;   // fragment elems 8..15 (VGPR 4..7)
  return u.v;
}

__device__ __forceinline__ v8f wmma_f16(v16h a, v16h b, v8f c) {
  return __builtin_amdgcn_wmma_f32_16x16x32_f16(false, a, false, b, (short)0, c,
                                                false, false);
}

// K=256 GEMM strip: A row base in LDS, B row in global f16 (row-major weights)
__device__ __forceinline__ v8f gemm_tile_256(const _Float16* abase,
                                             const _Float16* brow, int hi) {
  v8f acc = {};
#pragma unroll
  for (int ks = 0; ks < 8; ks++) {
    int ko = ks * 32;
    v16h a = ld_frag(abase + ko + 8 * hi, abase + ko + 16 + 8 * hi);
    v16h b = ld_frag(brow + ko + 8 * hi, brow + ko + 16 + 8 * hi);
    acc = wmma_f16(a, b, acc);
  }
  return acc;
}

// async global -> LDS, 16 bytes per lane (ASYNCcnt tracked)
__device__ __forceinline__ void async_g2l_b128(unsigned lds_byte_off,
                                               const void* gptr) {
  asm volatile("global_load_async_to_lds_b128 %0, %1, off"
               :: "v"(lds_byte_off), "v"(gptr) : "memory");
}

__device__ __forceinline__ void wait_asynccnt0() {
  asm volatile("s_wait_asynccnt 0x0" ::: "memory");
}

// ---- pre-pass: weights f32 -> f16 into workspace ----
__global__ __launch_bounds__(256) void convert_weights(
    const float* __restrict__ wqkv, const float* __restrict__ wout,
    _Float16* __restrict__ wqkv16, _Float16* __restrict__ wout16) {
  int i = blockIdx.x * 256 + threadIdx.x;
  if (i < 768 * 256) wqkv16[i] = (_Float16)wqkv[i];
  if (i < 256 * 256) wout16[i] = (_Float16)wout[i];
}

// ---- fused shifted-window attention: 1 block = 1 window (8x8 tokens) ----
__global__ __launch_bounds__(256) void swin_window_attn(
    const float* __restrict__ x,
    const _Float16* __restrict__ wqkv,    // (768,256) f16 row-major
    const float* __restrict__ bqkv,       // (768,)
    const _Float16* __restrict__ wout,    // (256,256) f16 row-major
    const float* __restrict__ bout,       // (256,)
    const float* __restrict__ penc,       // (8,225)
    float* __restrict__ out)              // (32,4096,256) f32
{
  extern __shared__ char smem[];
  _Float16* sX  = (_Float16*)(smem + OFF_X);
  _Float16* sQ  = (_Float16*)(smem + OFF_Q);
  _Float16* sK  = (_Float16*)(smem + OFF_K);
  _Float16* sVT = (_Float16*)(smem + OFF_VT);
  float*    sS  = (float*)(smem + OFF_S);
  _Float16* sO  = (_Float16*)(smem + OFF_O);
  float*    sRS = (float*)(smem + OFF_RS);
  float*    sPE = (float*)(smem + OFF_PE);
  int*      sRG = (int*)(smem + OFF_RG);

  const int tid  = threadIdx.x;
  const int lane = tid & 31;
  const int nlan = lane & 15;   // N / M index within a tile half
  const int hi   = lane >> 4;   // which lane-half
  const int wave = tid >> 5;    // 8 waves
  const int wIdx = blockIdx.x & 63;   // window index within image (8x8)
  const int img  = blockIdx.x >> 6;   // batch index
  const int wy = wIdx >> 3, wx = wIdx & 7;

  // ---------- stage 0a: async-copy x window (f32) into LDS staging ----------
  {
    int token = tid >> 2;
    int ly = token >> 3, lx = token & 7;
    int gy = (wy * 8 + ly + kShift) & 63;       // roll(-shift) folded in
    int gx = (wx * 8 + lx + kShift) & 63;
    const char* src = (const char*)(x + ((size_t)img * 4096 +
                                         (size_t)gy * 64 + gx) * kDim +
                                    (tid & 3) * 64);
    unsigned dst = (unsigned)(OFF_S + token * 1024 + (tid & 3) * 256);
#pragma unroll
    for (int i = 0; i < 16; i++)
      async_g2l_b128(dst + i * 16, src + i * 16);

    for (int i = tid; i < kHeads * 225; i += 256) sPE[i] = penc[i];
    if (tid < 64) {                              // region labels for the mask
      int gy2 = wy * 8 + (tid >> 3);
      int gx2 = wx * 8 + (tid & 7);
      int ry = (gy2 < 56) ? 0 : ((gy2 < 60) ? 1 : 2);
      int rx = (gx2 < 56) ? 0 : ((gx2 < 60) ? 1 : 2);
      sRG[tid] = ry * 3 + rx;
    }
    wait_asynccnt0();
  }
  __syncthreads();

  // ---------- stage 0b: staged f32 -> f16 ----------
  {
    const float* stage = (const float*)(smem + OFF_S);
#pragma unroll
    for (int i = 0; i < 64; i += 4) {
      int idx = tid * 64 + i;
      float4 v = *(const float4*)(stage + idx);
      sX[idx + 0] = (_Float16)v.x; sX[idx + 1] = (_Float16)v.y;
      sX[idx + 2] = (_Float16)v.z; sX[idx + 3] = (_Float16)v.w;
    }
  }
  __syncthreads();

  // ---------- stage 1: QKV projection  [64x256] x [256x768] ----------
  // Q tiles: features [0,256)
#pragma unroll
  for (int tt = 0; tt < 2; tt++) {
    int f = (wave + tt * 8) * 16 + nlan;
    const _Float16* brow = wqkv + (size_t)f * kDim;
    float bias = bqkv[f];
#pragma unroll
    for (int mt = 0; mt < 4; mt++) {
      v8f acc = gemm_tile_256(sX + (mt * 16 + nlan) * kDim, brow, hi);
#pragma unroll
      for (int r = 0; r < 8; r++) {
        int token = mt * 16 + r + 8 * hi;
        sQ[token * kDim + f] = (_Float16)((acc[r] + bias) * kScale);
      }
    }
  }
  // K tiles: features [256,512)
#pragma unroll
  for (int tt = 0; tt < 2; tt++) {
    int fk = (wave + tt * 8) * 16 + nlan;        // 0..255 within K
    const _Float16* brow = wqkv + (size_t)(256 + fk) * kDim;
    float bias = bqkv[256 + fk];
#pragma unroll
    for (int mt = 0; mt < 4; mt++) {
      v8f acc = gemm_tile_256(sX + (mt * 16 + nlan) * kDim, brow, hi);
#pragma unroll
      for (int r = 0; r < 8; r++) {
        int token = mt * 16 + r + 8 * hi;
        sK[token * kDim + fk] = (_Float16)(acc[r] + bias);
      }
    }
  }
  // V tiles: features [512,768) -> transposed [dim][token]
#pragma unroll
  for (int tt = 0; tt < 2; tt++) {
    int fv = (wave + tt * 8) * 16 + nlan;        // 0..255 within V
    const _Float16* brow = wqkv + (size_t)(512 + fv) * kDim;
    float bias = bqkv[512 + fv];
#pragma unroll
    for (int mt = 0; mt < 4; mt++) {
      v8f acc = gemm_tile_256(sX + (mt * 16 + nlan) * kDim, brow, hi);
#pragma unroll
      for (int r = 0; r < 8; r++) {
        int token = mt * 16 + r + 8 * hi;
        sVT[fv * 64 + token] = (_Float16)(acc[r] + bias);
      }
    }
  }
  __syncthreads();

  // ---------- stage 2: scores S = q k^T + bias (+mask), per head ----------
  const int h = wave;
  float* sSh = sS + h * 4096;              // 64x64 f32
#pragma unroll
  for (int mt = 0; mt < 4; mt++) {
    const _Float16* arow = sQ + (mt * 16 + nlan) * kDim + h * 32;
    v16h a = ld_frag(arow + 8 * hi, arow + 16 + 8 * hi);
#pragma unroll
    for (int nt = 0; nt < 4; nt++) {
      const _Float16* brow = sK + (nt * 16 + nlan) * kDim + h * 32;
      v16h b = ld_frag(brow + 8 * hi, brow + 16 + 8 * hi);
      v8f acc = {};
      acc = wmma_f16(a, b, acc);
      int kt = nt * 16 + nlan;
      int rk = sRG[kt];
      int y2 = kt >> 3, x2 = kt & 7;
#pragma unroll
      for (int r = 0; r < 8; r++) {
        int qt = mt * 16 + r + 8 * hi;
        int y1 = qt >> 3, x1 = qt & 7;
        int rel = (y1 - y2 + 7) * 15 + (x1 - x2 + 7);
        float sv = acc[r] + sPE[h * 225 + rel];
        if (sRG[qt] != rk) sv = -__builtin_inff();
        sSh[qt * 64 + kt] = sv;
      }
    }
  }
  // (intra-wave LDS ordering: this wave wrote all of sSh for its head)

  // ---------- stage 3: softmax (2 rows per lane); P f16 in place ----------
#pragma unroll
  for (int rr = 0; rr < 2; rr++) {
    int row = lane * 2 + rr;
    float* srow = sSh + row * 64;
    _Float16* prow = (_Float16*)srow;      // first 128B of the row
    float mx = -__builtin_inff();
#pragma unroll
    for (int j = 0; j < 64; j += 4) {
      float4 v = *(const float4*)(srow + j);
      mx = fmaxf(mx, fmaxf(fmaxf(v.x, v.y), fmaxf(v.z, v.w)));
    }
    float sum = 0.f;
#pragma unroll
    for (int j = 0; j < 64; j += 4) {
      float4 v = *(const float4*)(srow + j);   // read before overwrite
      float e0 = __expf(v.x - mx), e1 = __expf(v.y - mx);
      float e2 = __expf(v.z - mx), e3 = __expf(v.w - mx);
      sum += (e0 + e1) + (e2 + e3);
      prow[j + 0] = (_Float16)e0; prow[j + 1] = (_Float16)e1;
      prow[j + 2] = (_Float16)e2; prow[j + 3] = (_Float16)e3;
    }
    sRS[h * 64 + row] = sum;                 // 1/sum folded into P@V epilogue
  }

  // ---------- stage 4: O_h = (P/rowsum) @ V, per head ----------
  const _Float16* pbase = (const _Float16*)sSh;  // row stride 128 halfs
#pragma unroll
  for (int mt = 0; mt < 4; mt++) {
    const _Float16* arow = pbase + (mt * 16 + nlan) * 128;
#pragma unroll
    for (int nt = 0; nt < 2; nt++) {
      const _Float16* brow = sVT + (h * 32 + nt * 16 + nlan) * 64;
      v8f acc = {};
#pragma unroll
      for (int ks = 0; ks < 2; ks++) {
        v16h a = ld_frag(arow + ks * 32 + 8 * hi, arow + ks * 32 + 16 + 8 * hi);
        v16h b = ld_frag(brow + ks * 32 + 8 * hi, brow + ks * 32 + 16 + 8 * hi);
        acc = wmma_f16(a, b, acc);
      }
#pragma unroll
      for (int r = 0; r < 8; r++) {
        int token = mt * 16 + r + 8 * hi;
        float inv = 1.0f / sRS[h * 64 + token];
        sO[token * kDim + h * 32 + nt * 16 + nlan] = (_Float16)(acc[r] * inv);
      }
    }
  }
  __syncthreads();

  // ---------- stage 5: out projection + bias, store with roll(+shift) ----------
#pragma unroll
  for (int tt = 0; tt < 2; tt++) {
    int ch = (wave + tt * 8) * 16 + nlan;
    const _Float16* brow = wout + (size_t)ch * kDim;
    float bias = bout[ch];
#pragma unroll
    for (int mt = 0; mt < 4; mt++) {
      v8f acc = gemm_tile_256(sO + (mt * 16 + nlan) * kDim, brow, hi);
#pragma unroll
      for (int r = 0; r < 8; r++) {
        int token = mt * 16 + r + 8 * hi;
        int ly = token >> 3, lx = token & 7;
        int dy = (wy * 8 + ly + kShift) & 63;
        int dx = (wx * 8 + lx + kShift) & 63;
        out[((size_t)img * 4096 + (size_t)dy * 64 + dx) * kDim + ch] =
            acc[r] + bias;
      }
    }
  }
}

extern "C" void kernel_launch(void* const* d_in, const int* in_sizes, int n_in,
                              void* d_out, int out_size, void* d_ws, size_t ws_size,
                              hipStream_t stream) {
  const float* x    = (const float*)d_in[0];
  const float* wqkv = (const float*)d_in[1];
  const float* bqkv = (const float*)d_in[2];
  const float* wout = (const float*)d_in[3];
  const float* bout = (const float*)d_in[4];
  const float* penc = (const float*)d_in[5];
  float* out = (float*)d_out;

  _Float16* wqkv16 = (_Float16*)d_ws;
  _Float16* wout16 = (_Float16*)((char*)d_ws + (size_t)768 * 256 * 2);

  convert_weights<<<768, 256, 0, stream>>>(wqkv, wout, wqkv16, wout16);

  (void)hipFuncSetAttribute((const void*)swin_window_attn,
                            hipFuncAttributeMaxDynamicSharedMemorySize,
                            kSmemBytes);
  // 32 images * 64 windows = 2048 blocks, 8 waves each
  swin_window_attn<<<2048, 256, kSmemBytes, stream>>>(
      x, wqkv16, bqkv, wout16, bout, penc, out);
}
